// AttendAndSpell_59863254172710
// MI455X (gfx1250) — matrix operand order repaired and checked by reading, not compiled, not run on hardware
//
#include <hip/hip_runtime.h>
#include <hip/hip_bf16.h>

// Attend-and-Spell decoder for MI455X (gfx1250, wave32, WMMA).
// B=64, R=256, T=128, H=512, V=4096.
//
// All dense GEMMs: v_wmma_f32_16x16x32_bf16 (bf16 A/B, f32 accum).
// Weights AND activations are kept in bf16 buffers so GEMM inner loops are
// pure {2x b128 A-load, 2x b128 B-load, wmma}; f32 only for cell state,
// softmax, biases and the final logits.

#define B_  64
#define R_  256
#define T_  128
#define H_  512
#define V_  4096

typedef __attribute__((ext_vector_type(16))) __bf16 v16bf;
typedef __attribute__((ext_vector_type(8)))  float  v8f;
typedef unsigned short u16t;

union BF16Frag {
    v16bf v;
    u16t  u[16];
    uint4 q[2];
};

static __device__ __forceinline__ u16t f2bf(float x) {
    unsigned int u = __float_as_uint(x);
    unsigned int r = (u + 0x7FFFu + ((u >> 16) & 1u)) >> 16;  // RNE
    return (u16t)r;
}

// ---- A fragment: 16x32 bf16 tile from row-major bf16 activations [M,K] ----
// Lane L holds row m0+(L&15); K = k0 + (L>=16?16:0) + 0..15 (contiguous).
static __device__ __forceinline__ v16bf loadA_bf16(const u16t* __restrict__ X,
                                                   int lda, int m0, int k0,
                                                   int lane) {
    int m = m0 + (lane & 15);
    int half = lane >> 4;
    const u16t* p = X + (size_t)m * lda + k0 + half * 16;
    BF16Frag f;
    f.q[0] = *(const uint4*)(p);        // 8 bf16, K run 1
    f.q[1] = *(const uint4*)(p + 8);    // 8 bf16, K run 2
    return f.v;
}

// ---- B fragment: 32x16 bf16 tile from row-major bf16 weights [N,K] --------
// Lane L holds column n0+(L&15); K = k0 + (L>=16?16:0) + 0..15 (contiguous).
static __device__ __forceinline__ v16bf loadB_bf16(const u16t* __restrict__ W,
                                                   int ldb, int n0, int k0,
                                                   int lane) {
    int n = n0 + (lane & 15);
    int half = lane >> 4;
    const u16t* p = W + (size_t)n * ldb + k0 + half * 16;
    BF16Frag f;
    f.q[0] = *(const uint4*)(p);
    f.q[1] = *(const uint4*)(p + 8);
    return f.v;
}

static __device__ __forceinline__ v8f wmma_bf16(v16bf a, v16bf b, v8f c) {
    return __builtin_amdgcn_wmma_f32_16x16x32_bf16(false, a, false, b,
                                                   (short)0, c, false, false);
}

// ---------------------------------------------------------------------------
// GEMM: C[M,N] = A[M,K](bf16) * W[N,K](bf16)^T + bias -> f32.
// One wave per 64(M) x 16(N) strip: each B fragment feeds 4 WMMAs.
// grid = (M/64, N/16), block = 32.
// ---------------------------------------------------------------------------
__global__ __launch_bounds__(32)
void wmma_gemm_bias(const u16t* __restrict__ A, int lda,
                    const u16t* __restrict__ W,
                    const float* __restrict__ bias,
                    float* __restrict__ C, long long ldc, int K) {
    int lane = threadIdx.x & 31;
    int m0 = blockIdx.x * 64;
    int n0 = blockIdx.y * 16;
    v8f acc[4] = {};
    for (int k0 = 0; k0 < K; k0 += 32) {
        v16bf b = loadB_bf16(W, K, n0, k0, lane);
        if (k0 + 32 < K) {   // prefetch next B rows (global_prefetch_b8)
            __builtin_prefetch(W + (size_t)(n0 + (lane & 15)) * K + k0 + 32, 0, 1);
        }
#pragma unroll
        for (int mt = 0; mt < 4; ++mt) {
            v16bf a = loadA_bf16(A, lda, m0 + mt * 16, k0, lane);
            acc[mt] = wmma_bf16(a, b, acc[mt]);
        }
    }
    int col = n0 + (lane & 15);
    int half = lane >> 4;
    float bv = bias[col];
#pragma unroll
    for (int mt = 0; mt < 4; ++mt) {
#pragma unroll
        for (int v = 0; v < 8; ++v) {
            int row = m0 + mt * 16 + v + half * 8;  // C layout: VGPR v -> M
            C[(size_t)row * ldc + col] = acc[mt][v] + bv;
        }
    }
}

// ---------------------------------------------------------------------------
// Fused LSTM cell.  Wave computes the 4 gate tiles (i,f,g,o) for one
// 16(batch) x 16(hidden) slice (A fragment shared across gates), then the
// cell update.  States are bf16 except cs (f32).  grid = (B/16, H/16).
// Wgather folds in the one-hot column gather w_ih0[:, z_b] (f32).
// ---------------------------------------------------------------------------
__global__ __launch_bounds__(32)
void lstm_cell_kernel(const u16t*  __restrict__ X,      // [B,H] bf16
                      const u16t*  __restrict__ Hprev,  // [B,H] bf16
                      const u16t*  __restrict__ Wx,     // [4H,H] bf16
                      const u16t*  __restrict__ Wh,     // [4H,H] bf16
                      const float* __restrict__ b1,     // [4H]
                      const float* __restrict__ b2,     // [4H]
                      const float* __restrict__ Wgather,// [4H,H+V] f32 or null
                      const int*   __restrict__ y,      // [B,T] or null
                      int t,
                      float* __restrict__ cs,           // [B,H] f32 in/out
                      u16t*  __restrict__ s_out,        // [B,H] bf16
                      u16t*  __restrict__ extra)        // [B,2H] bf16 lo, or null
{
    int lane = threadIdx.x & 31;
    int m0 = blockIdx.x * 16;
    int n0 = blockIdx.y * 16;
    v8f acc[4] = {};

    for (int k0 = 0; k0 < H_; k0 += 32) {
        v16bf a = loadA_bf16(X, H_, m0, k0, lane);
#pragma unroll
        for (int g = 0; g < 4; ++g) {
            v16bf b = loadB_bf16(Wx, H_, g * H_ + n0, k0, lane);
            acc[g] = wmma_bf16(a, b, acc[g]);
        }
    }
    for (int k0 = 0; k0 < H_; k0 += 32) {
        v16bf a = loadA_bf16(Hprev, H_, m0, k0, lane);
#pragma unroll
        for (int g = 0; g < 4; ++g) {
            v16bf b = loadB_bf16(Wh, H_, g * H_ + n0, k0, lane);
            acc[g] = wmma_bf16(a, b, acc[g]);
        }
    }

    int col = n0 + (lane & 15);
    int half = lane >> 4;
    float bb[4];
#pragma unroll
    for (int g = 0; g < 4; ++g) bb[g] = b1[g * H_ + col] + b2[g * H_ + col];

#pragma unroll
    for (int v = 0; v < 8; ++v) {
        int row = m0 + v + half * 8;
        float pre[4];
#pragma unroll
        for (int g = 0; g < 4; ++g) pre[g] = acc[g][v] + bb[g];
        if (Wgather) {
            int z = y[row * T_ + t];
#pragma unroll
            for (int g = 0; g < 4; ++g)
                pre[g] += Wgather[(size_t)(g * H_ + col) * (H_ + V_) + z];
        }
        float ig = 1.f / (1.f + __expf(-pre[0]));
        float fg = 1.f / (1.f + __expf(-pre[1]));
        float gg = tanhf(pre[2]);
        float og = 1.f / (1.f + __expf(-pre[3]));
        size_t idx = (size_t)row * H_ + col;
        float cval = fg * cs[idx] + ig * gg;
        cs[idx] = cval;
        u16t s = f2bf(og * tanhf(cval));
        s_out[idx] = s;
        if (extra) extra[(size_t)row * (2 * H_) + col] = s;
    }
}

// ---------------------------------------------------------------------------
// Attention: e_r = <sp_b, hp[b,r,:]>/sqrt(H); softmax over r;
// c_b = sum_r alpha_r hp[b,r,:].  grid = B, block = 256 (= R).
// Emits c as bf16 (consumed as A fragments) into c_bf and s1c hi half.
// ---------------------------------------------------------------------------
__global__ __launch_bounds__(256)
void attend_kernel(const float* __restrict__ sp,  // [B,H] f32
                   const float* __restrict__ hp,  // [B,R,H] f32
                   u16t* __restrict__ c_bf,       // [B,H] bf16
                   u16t* __restrict__ s1c)        // [B,2H] bf16 hi half
{
    __shared__ float sp_s[H_];
    __shared__ float prob[R_];
    __shared__ float red[R_];
    int b = blockIdx.x;
    int tid = threadIdx.x;

    for (int d = tid; d < H_; d += 256) sp_s[d] = sp[b * H_ + d];
    __syncthreads();

    const float* hrow = hp + ((size_t)b * R_ + tid) * H_;
    float acc = 0.f;
    for (int d = 0; d < H_; d += 4) {
        acc += sp_s[d] * hrow[d] + sp_s[d + 1] * hrow[d + 1] +
               sp_s[d + 2] * hrow[d + 2] + sp_s[d + 3] * hrow[d + 3];
    }
    float e = acc * 0.044194173824159216f;  // 1/sqrt(512)

    red[tid] = e;
    __syncthreads();
    for (int s = 128; s > 0; s >>= 1) {
        if (tid < s) red[tid] = fmaxf(red[tid], red[tid + s]);
        __syncthreads();
    }
    float m = red[0];
    __syncthreads();
    float p = __expf(e - m);
    prob[tid] = p;
    red[tid] = p;
    __syncthreads();
    for (int s = 128; s > 0; s >>= 1) {
        if (tid < s) red[tid] += red[tid + s];
        __syncthreads();
    }
    float inv = 1.f / red[0];
    __syncthreads();

    float c0 = 0.f, c1 = 0.f;
    const float* hb = hp + (size_t)b * R_ * H_;
    for (int r = 0; r < R_; ++r) {
        float a = prob[r];
        c0 += a * hb[(size_t)r * H_ + tid];
        c1 += a * hb[(size_t)r * H_ + tid + 256];
    }
    u16t v0 = f2bf(c0 * inv);
    u16t v1 = f2bf(c1 * inv);
    c_bf[b * H_ + tid]       = v0;
    c_bf[b * H_ + tid + 256] = v1;
    s1c[(size_t)b * (2 * H_) + H_ + tid]       = v0;
    s1c[(size_t)b * (2 * H_) + H_ + tid + 256] = v1;
}

// ---------------------------------------------------------------------------
// Setup kernels
// ---------------------------------------------------------------------------
__global__ void conv_bf16(const float* __restrict__ src, u16t* __restrict__ dst,
                          int n) {
    int i = blockIdx.x * blockDim.x + threadIdx.x;
    if (i < n) dst[i] = f2bf(src[i]);
}

// Extract columns V..V+H-1 of w_ih0 ([4H, H+V]) as dense bf16 [4H, H].
__global__ void conv_wih0c(const float* __restrict__ w, u16t* __restrict__ dst) {
    int i = blockIdx.x * blockDim.x + threadIdx.x;
    if (i < 4 * H_ * H_) {
        int row = i >> 9;       // /H_
        int col = i & (H_ - 1);
        dst[i] = f2bf(w[(size_t)row * (H_ + V_) + V_ + col]);
    }
}

__global__ void init_state(u16t* s0, float* cs0, u16t* s1, float* cs1,
                           float* sp, const float* __restrict__ phi_b) {
    int i = blockIdx.x * blockDim.x + threadIdx.x;
    if (i < B_ * H_) {
        s0[i] = 0; cs0[i] = 0.f; s1[i] = 0; cs1[i] = 0.f;
        sp[i] = phi_b[i & (H_ - 1)];  // phi(0) = phi_b
    }
}

// ---------------------------------------------------------------------------
extern "C" void kernel_launch(void* const* d_in, const int* in_sizes, int n_in,
                              void* d_out, int out_size, void* d_ws,
                              size_t ws_size, hipStream_t stream) {
    const float* h      = (const float*)d_in[0];   // [B,R,H]
    const int*   y      = (const int*)  d_in[1];   // [B,T]
    const float* phi_w  = (const float*)d_in[3];
    const float* phi_b  = (const float*)d_in[4];
    const float* psi_w  = (const float*)d_in[5];
    const float* psi_b  = (const float*)d_in[6];
    const float* w_ih0  = (const float*)d_in[7];   // [2048,4608]
    const float* w_hh0  = (const float*)d_in[8];
    const float* b_ih0  = (const float*)d_in[9];
    const float* b_hh0  = (const float*)d_in[10];
    const float* w_ih1  = (const float*)d_in[11];
    const float* w_hh1  = (const float*)d_in[12];
    const float* b_ih1  = (const float*)d_in[13];
    const float* b_hh1  = (const float*)d_in[14];
    const float* out_w  = (const float*)d_in[15];  // [4096,1024]
    const float* out_b  = (const float*)d_in[16];
    float* out = (float*)d_out;

    // ---- workspace layout (256B aligned slices) ----
    char* base = (char*)d_ws;
    size_t off = 0;
    auto alloc = [&](size_t bytes) -> char* {
        char* p = base + off;
        off = (off + bytes + 255) & ~(size_t)255;
        return p;
    };
    float* hp      = (float*)alloc((size_t)B_ * R_ * H_ * 4);   // 33.5 MB
    u16t* h_bf     = (u16t*)alloc((size_t)B_ * R_ * H_ * 2);    // 16.8 MB
    u16t* psi_bf   = (u16t*)alloc((size_t)H_ * H_ * 2);
    u16t* phi_bf   = (u16t*)alloc((size_t)H_ * H_ * 2);
    u16t* wih0c_bf = (u16t*)alloc((size_t)4 * H_ * H_ * 2);
    u16t* whh0_bf  = (u16t*)alloc((size_t)4 * H_ * H_ * 2);
    u16t* wih1_bf  = (u16t*)alloc((size_t)4 * H_ * H_ * 2);
    u16t* whh1_bf  = (u16t*)alloc((size_t)4 * H_ * H_ * 2);
    u16t* outw_bf  = (u16t*)alloc((size_t)V_ * 2 * H_ * 2);     // 8.4 MB
    u16t* s0p[2]   = {(u16t*)alloc(B_ * H_ * 2), (u16t*)alloc(B_ * H_ * 2)};
    float* cs0     = (float*)alloc(B_ * H_ * 4);
    u16t* s1p[2]   = {(u16t*)alloc(B_ * H_ * 2), (u16t*)alloc(B_ * H_ * 2)};
    float* cs1     = (float*)alloc(B_ * H_ * 4);
    u16t* c_bf     = (u16t*)alloc(B_ * H_ * 2);
    float* sp      = (float*)alloc(B_ * H_ * 4);
    u16t* s1c      = (u16t*)alloc((size_t)B_ * 2 * H_ * 2);     // [s1 | c] bf16

    // ---- one-time conversions to bf16 ----
    auto cv = [&](const float* src, u16t* dst, int n) {
        conv_bf16<<<(n + 255) / 256, 256, 0, stream>>>(src, dst, n);
    };
    cv(psi_w, psi_bf, H_ * H_);
    cv(phi_w, phi_bf, H_ * H_);
    cv(w_hh0, whh0_bf, 4 * H_ * H_);
    cv(w_ih1, wih1_bf, 4 * H_ * H_);
    cv(w_hh1, whh1_bf, 4 * H_ * H_);
    cv(out_w, outw_bf, V_ * 2 * H_);
    cv(h, h_bf, B_ * R_ * H_);
    conv_wih0c<<<(4 * H_ * H_ + 255) / 256, 256, 0, stream>>>(w_ih0, wih0c_bf);

    // ---- hp = h @ psi_w^T + psi_b   ([16384,512] x [512,512]) ----
    wmma_gemm_bias<<<dim3((B_ * R_) / 64, H_ / 16), 32, 0, stream>>>(
        h_bf, H_, psi_bf, psi_b, hp, H_, H_);

    // ---- init: states = 0, sp = phi(0) = phi_b; c = attend(0) ----
    init_state<<<(B_ * H_ + 255) / 256, 256, 0, stream>>>(
        s0p[0], cs0, s1p[0], cs1, sp, phi_b);
    attend_kernel<<<B_, 256, 0, stream>>>(sp, hp, c_bf, s1c);

    // ---- recurrence ----
    for (int t = 0; t < T_; ++t) {
        u16t* s0in  = s0p[t & 1];
        u16t* s0out = s0p[(t & 1) ^ 1];
        u16t* s1in  = s1p[t & 1];
        u16t* s1out = s1p[(t & 1) ^ 1];

        // LSTM cell 0: x = [onehot(z) | c]
        lstm_cell_kernel<<<dim3(B_ / 16, H_ / 16), 32, 0, stream>>>(
            c_bf, s0in, wih0c_bf, whh0_bf, b_ih0, b_hh0,
            w_ih0, y, t, cs0, s0out, nullptr);

        // LSTM cell 1: x = s0  (also writes s1 into lo half of [s1|c])
        lstm_cell_kernel<<<dim3(B_ / 16, H_ / 16), 32, 0, stream>>>(
            s0out, s1in, wih1_bf, whh1_bf, b_ih1, b_hh1,
            nullptr, nullptr, 0, cs1, s1out, s1c);

        // sp = s1 @ phi_w^T + phi_b   (A = lo half of s1c, lda = 2H)
        wmma_gemm_bias<<<dim3(1, H_ / 16), 32, 0, stream>>>(
            s1c, 2 * H_, phi_bf, phi_b, sp, H_, H_);

        // c = attend(s1)  (writes c_bf and hi half of s1c)
        attend_kernel<<<B_, 256, 0, stream>>>(sp, hp, c_bf, s1c);

        // o_t = [s1|c] @ out_w^T + out_b  -> d_out[:, t, :]
        wmma_gemm_bias<<<dim3(1, V_ / 16), 32, 0, stream>>>(
            s1c, 2 * H_, outw_bf, out_b, out + (size_t)t * V_,
            (long long)T_ * V_, 2 * H_);
    }
}
